// AttentionLSTM_Multiplicative_28192165331639
// MI455X (gfx1250) — matrix operand order repaired and checked by reading, not compiled, not run on hardware
//
#include <hip/hip_runtime.h>
#include <hip/hip_bf16.h>
#include <math.h>

// ---------------------------------------------------------------------------
// Types for CDNA5 WMMA (wave32): v_wmma_f32_16x16x32_bf16
// ---------------------------------------------------------------------------
typedef __bf16 v16bf __attribute__((ext_vector_type(16)));
typedef __bf16 v8bf  __attribute__((ext_vector_type(8)));
typedef __bf16 v4bf  __attribute__((ext_vector_type(4)));
typedef float  v8f   __attribute__((ext_vector_type(8)));

#define BSZ   64
#define TLEN  512
#define EDIM  256
#define HDIM  512
#define KTOT  768   // E + H fused K dimension

__device__ __forceinline__ float sigf(float x) { return 1.0f / (1.0f + __expf(-x)); }

// One 16x16x32 bf16 WMMA accumulate.
// A 16x32 bf16 layout (ISA 7.12.2): lane lr holds row M=lr;
//   half==0 lanes: a[0..7]=K 0..7,  a[8..15]=K 16..23
//   half==1 lanes: a[0..7]=K 8..15, a[8..15]=K 24..31
// B 32x16 bf16 layout: lane lr holds col N=lr; same K order as A
__device__ __forceinline__
v8f tile_mma(const __bf16* arow, const __bf16* wrow, v8f acc, int half) {
    v8bf alo = *(const v8bf*)(arow + half * 8);
    v8bf ahi = *(const v8bf*)(arow + 16 + half * 8);
    v16bf af;
#pragma unroll
    for (int i = 0; i < 8; ++i) { af[i] = alo[i]; af[i + 8] = ahi[i]; }
    v8bf blo = *(const v8bf*)(wrow + half * 8);
    v8bf bhi = *(const v8bf*)(wrow + 16 + half * 8);
    v16bf bfm;
#pragma unroll
    for (int i = 0; i < 8; ++i) { bfm[i] = blo[i]; bfm[i + 8] = bhi[i]; }
    return __builtin_amdgcn_wmma_f32_16x16x32_bf16(
        /*neg_a=*/false, af, /*neg_b=*/false, bfm,
        /*c_mod=*/(short)0, acc, /*reuse_a=*/false, /*reuse_b=*/false);
}

// ---------------------------------------------------------------------------
// k_prep: fuse the 4 input + 4 recurrent weight matrices into bf16
//         Wc[g][n][k] (k<256 from Wi[n][k], k>=256 from Wh[n][k-256]);
//         fold the two bias vectors per gate into bc[g][n].
// ---------------------------------------------------------------------------
__global__ void k_prep(const float* __restrict__ Wir, const float* __restrict__ Wif,
                       const float* __restrict__ Wig, const float* __restrict__ Wio,
                       const float* __restrict__ Whr, const float* __restrict__ Whf,
                       const float* __restrict__ Whg, const float* __restrict__ Who,
                       const float* __restrict__ bir, const float* __restrict__ bif,
                       const float* __restrict__ big_, const float* __restrict__ bio,
                       const float* __restrict__ bhr, const float* __restrict__ bhf,
                       const float* __restrict__ bhg, const float* __restrict__ bho,
                       __bf16* __restrict__ Wc, float* __restrict__ bc) {
    int i = blockIdx.x * blockDim.x + threadIdx.x;
    const int tot = 4 * HDIM * KTOT;
    if (i < tot) {
        int g = i / (HDIM * KTOT);
        int rem = i % (HDIM * KTOT);
        int n = rem / KTOT;
        int k = rem % KTOT;
        const float* Wi[4] = {Wir, Wif, Wig, Wio};
        const float* Wh[4] = {Whr, Whf, Whg, Who};
        float v = (k < EDIM) ? Wi[g][n * EDIM + k] : Wh[g][n * HDIM + (k - EDIM)];
        Wc[i] = (__bf16)v;
    }
    if (i < 4 * HDIM) {
        const float* bi[4] = {bir, bif, big_, bio};
        const float* bh[4] = {bhr, bhf, bhg, bho};
        int g = i >> 9, n = i & (HDIM - 1);
        bc[i] = bi[g][n] + bh[g][n];
    }
}

// ---------------------------------------------------------------------------
// lstm_scan: one persistent workgroup (1024 thr = 32 wave32 waves).
// Per step: vectorized embedding gather -> bf16 LDS, WMMA over K=768 for the
// 4 gates (sequentially, to keep VGPR pressure low; c-state lives in VGPRs),
// h written bf16 into double-buffered LDS, then the whole 64KB h-tile is
// streamed to the global hs history with global_store_async_from_lds_b128.
// ---------------------------------------------------------------------------
__global__ __launch_bounds__(1024)
void lstm_scan(const int* __restrict__ words, const float* __restrict__ emb,
               const __bf16* __restrict__ Wc, const float* __restrict__ bc,
               __bf16* __restrict__ hs, float* __restrict__ hlast,
               float* __restrict__ hlast_out) {
    extern __shared__ char smem[];
    const unsigned AOFF = BSZ * EDIM * 2;            // hbufA LDS byte offset
    const unsigned BOFF = AOFF + BSZ * HDIM * 2;     // hbufB LDS byte offset
    __bf16* xbuf  = (__bf16*)smem;                   // [64][256]
    __bf16* hbufA = (__bf16*)(smem + AOFF);          // [64][512]
    __bf16* hbufB = (__bf16*)(smem + BOFF);          // [64][512]

    const int tid  = threadIdx.x;
    const int wave = tid >> 5;
    const int lane = tid & 31;
    const int half = lane >> 4;
    const int lr   = lane & 15;

    for (int i = tid; i < BSZ * HDIM; i += 1024) {
        hbufA[i] = (__bf16)0.0f;
        hbufB[i] = (__bf16)0.0f;
    }

    // persistent per-wave state: c for the 4 owned (m-tile, h-col-tile) positions
    float creg[4][8];
#pragma unroll
    for (int p = 0; p < 4; ++p)
#pragma unroll
        for (int v = 0; v < 8; ++v) creg[p][v] = 0.0f;

    // t-invariant fused biases for this wave's positions
    float biasreg[4][4];
#pragma unroll
    for (int p = 0; p < 4; ++p) {
        int h0 = ((wave * 4 + p) >> 2) * 16;
#pragma unroll
        for (int g = 0; g < 4; ++g) biasreg[p][g] = bc[g * HDIM + h0 + lr];
    }

    __syncthreads();

    for (int t = 0; t < TLEN; ++t) {
        const __bf16* hcur     = (t & 1) ? hbufB : hbufA;
        __bf16*       hnxt     = (t & 1) ? hbufA : hbufB;
        const unsigned hnxt_off = (t & 1) ? AOFF : BOFF;

        // gather x_t = emb[words[:, t]] -> bf16 LDS (float4 -> 4 x bf16)
        for (int j = tid; j < (BSZ * EDIM) / 4; j += 1024) {
            int b = j >> 6;              // 64 float4 per batch row
            int e = (j & 63) * 4;
            int wd = words[b * TLEN + t];
            const float4 f = *(const float4*)(emb + (size_t)wd * EDIM + e);
            v4bf o;
            o[0] = (__bf16)f.x; o[1] = (__bf16)f.y; o[2] = (__bf16)f.z; o[3] = (__bf16)f.w;
            *(v4bf*)(xbuf + b * EDIM + e) = o;
        }
        // drain this wave's async stores from step t-2 (they read the buffer we
        // are about to overwrite as hnxt); barrier below makes this global.
        asm volatile("s_wait_asynccnt 0x4" ::: "memory");
        __syncthreads();

#pragma unroll
        for (int p = 0; p < 4; ++p) {
            const int pos = wave * 4 + p;      // 0..127
            const int m0  = (pos & 3) * 16;    // batch tile
            const int h0  = (pos >> 2) * 16;   // hidden-col tile
            const __bf16* arow_x = xbuf + (m0 + lr) * EDIM;
            const __bf16* arow_h = hcur + (m0 + lr) * HDIM;

            // one gate pre-activation: bias + [x_t | h] @ [Wi|Wh]^T over K=768
            auto gate = [&](int g) -> v8f {
                const float bias = biasreg[p][g];
                v8f a = {bias, bias, bias, bias, bias, bias, bias, bias};
                const __bf16* wrow = Wc + ((size_t)(g * HDIM + h0 + lr)) * KTOT;
                __builtin_prefetch(wrow, 0, 1);
                __builtin_prefetch(wrow + 512, 0, 1);
#pragma unroll 4
                for (int kk = 0; kk < 8; ++kk)          // x part (K=256)
                    a = tile_mma(arow_x + kk * 32, wrow + kk * 32, a, half);
#pragma unroll 4
                for (int kk = 0; kk < 16; ++kk)         // recurrent part (K=512)
                    a = tile_mma(arow_h + kk * 32, wrow + EDIM + kk * 32, a, half);
                return a;
            };

            // sequential-gate combine keeps at most one accumulator live
            v8f aR = gate(0);
            float rg[8];
#pragma unroll
            for (int v = 0; v < 8; ++v) rg[v] = sigf(aR[v]);
            v8f aF = gate(1);
#pragma unroll
            for (int v = 0; v < 8; ++v) creg[p][v] *= sigf(aF[v]);
            v8f aG = gate(2);
#pragma unroll
            for (int v = 0; v < 8; ++v) creg[p][v] += rg[v] * tanhf(aG[v]);
            v8f aO = gate(3);
#pragma unroll
            for (int v = 0; v < 8; ++v) {
                float h = sigf(aO[v]) * tanhf(creg[p][v]);
                int m = m0 + v + half * 8;     // D layout: M = v (+8 for hi lanes)
                hnxt[m * HDIM + h0 + lr] = (__bf16)h;
            }
        }
        __syncthreads();

        // stream hnxt (== hs[t], 64KB bf16) to global via async LDS->mem path
        {
            const size_t gbase = (size_t)t * BSZ * HDIM * 2;
#pragma unroll
            for (int j2 = 0; j2 < 4; ++j2) {
                unsigned idx  = (unsigned)tid + j2 * 1024u;   // 16B chunks
                unsigned loff = hnxt_off + idx * 16u;
                const char* gp = (const char*)hs + gbase + (size_t)idx * 16u;
                asm volatile("global_store_async_from_lds_b128 %0, %1, off"
                             :: "v"(gp), "v"(loff) : "memory");
            }
        }
    }

    // final h lives in hbufA (t=511 is odd -> hnxt was hbufA)
    __syncthreads();
    for (int i = tid; i < BSZ * HDIM; i += 1024) {
        float h = (float)hbufA[i];
        hlast[i] = h;
        hlast_out[i] = h;
    }
}

// ---------------------------------------------------------------------------
// Attention / output epilogue (tiny: ~0.1 GFLOP total)
// ---------------------------------------------------------------------------
__global__ void k_fh(const float* __restrict__ hlast, const float* __restrict__ W_ol,
                     const float* __restrict__ b_ol, float* __restrict__ fh) {
    int i = blockIdx.x * blockDim.x + threadIdx.x;
    if (i >= BSZ * 256) return;
    int b = i >> 8, j = i & 255;
    const float* hr = hlast + b * HDIM;
    const float* wr = W_ol + j * HDIM;
    float s = b_ol[j];
    for (int k = 0; k < HDIM; ++k) s += hr[k] * wr[k];
    fh[i] = s;
}

__global__ void k_score(const float* __restrict__ fh, const float* __restrict__ W_att,
                        const float* __restrict__ b_att, float* __restrict__ score) {
    int i = blockIdx.x * blockDim.x + threadIdx.x;
    if (i >= BSZ * HDIM) return;
    int b = i >> 9, h = i & (HDIM - 1);
    float s = b_att[h];
    const float* fr = fh + b * 256;
    for (int j = 0; j < 256; ++j) s += fr[j] * W_att[j * HDIM + h];
    score[i] = s;
}

__global__ void k_attsc(const float* __restrict__ score, const __bf16* __restrict__ hs,
                        float* __restrict__ att) {
    int i = blockIdx.x * blockDim.x + threadIdx.x;
    if (i >= BSZ * TLEN) return;
    int b = i >> 9, t = i & (TLEN - 1);
    const float*  sr = score + b * HDIM;
    const __bf16* hr = hs + ((size_t)t * BSZ + b) * HDIM;
    float s = 0.0f;
    for (int h = 0; h < HDIM; ++h) s += sr[h] * (float)hr[h];
    att[i] = s;
}

__global__ void k_softmax(const float* __restrict__ att, float* __restrict__ dist) {
    __shared__ float red[256];
    int b = blockIdx.x, tid = threadIdx.x;
    float mx = -1e30f;
    for (int t = tid; t < TLEN; t += 256) mx = fmaxf(mx, att[b * TLEN + t]);
    red[tid] = mx; __syncthreads();
    for (int s = 128; s > 0; s >>= 1) { if (tid < s) red[tid] = fmaxf(red[tid], red[tid + s]); __syncthreads(); }
    mx = red[0]; __syncthreads();
    float sum = 0.0f;
    for (int t = tid; t < TLEN; t += 256) {
        float e = __expf(att[b * TLEN + t] - mx);
        dist[b * TLEN + t] = e;
        sum += e;
    }
    red[tid] = sum; __syncthreads();
    for (int s = 128; s > 0; s >>= 1) { if (tid < s) red[tid] += red[tid + s]; __syncthreads(); }
    float inv = 1.0f / red[0];
    for (int t = tid; t < TLEN; t += 256) dist[b * TLEN + t] *= inv;
}

__global__ void k_attout(const __bf16* __restrict__ hs, const float* __restrict__ dist,
                         float* __restrict__ attout) {
    int i = blockIdx.x * blockDim.x + threadIdx.x;
    if (i >= BSZ * HDIM) return;
    int b = i >> 9, h = i & (HDIM - 1);
    float s = 0.0f;
    for (int t = 0; t < TLEN; ++t)
        s += (float)hs[((size_t)t * BSZ + b) * HDIM + h] * dist[b * TLEN + t];
    attout[i] = s;
}

__global__ void k_out(const float* __restrict__ fh, const float* __restrict__ attout,
                      const float* __restrict__ W_fc, const float* __restrict__ b_fc,
                      float* __restrict__ out) {
    int i = blockIdx.x * blockDim.x + threadIdx.x;
    if (i >= BSZ * 2) return;
    int b = i >> 1, o = i & 1;
    const float* wr = W_fc + o * (256 + HDIM);
    float z = b_fc[o];
    for (int j = 0; j < 256; ++j)  z += fh[b * 256 + j] * wr[j];
    for (int h = 0; h < HDIM; ++h) z += attout[b * HDIM + h] * wr[256 + h];
    out[i] = sigf(z);
}

// ---------------------------------------------------------------------------
// Host-side launcher
// ---------------------------------------------------------------------------
extern "C" void kernel_launch(void* const* d_in, const int* in_sizes, int n_in,
                              void* d_out, int out_size, void* d_ws, size_t ws_size,
                              hipStream_t stream) {
    const int*   input_words = (const int*)d_in[0];
    const float* emb   = (const float*)d_in[1];
    const float* Wir   = (const float*)d_in[2];  const float* bir  = (const float*)d_in[3];
    const float* Whr   = (const float*)d_in[4];  const float* bhr  = (const float*)d_in[5];
    const float* Wif   = (const float*)d_in[6];  const float* bif  = (const float*)d_in[7];
    const float* Whf   = (const float*)d_in[8];  const float* bhf  = (const float*)d_in[9];
    const float* Wig   = (const float*)d_in[10]; const float* big_ = (const float*)d_in[11];
    const float* Whg   = (const float*)d_in[12]; const float* bhg  = (const float*)d_in[13];
    const float* Wio   = (const float*)d_in[14]; const float* bio  = (const float*)d_in[15];
    const float* Who   = (const float*)d_in[16]; const float* bho  = (const float*)d_in[17];
    const float* W_att = (const float*)d_in[18]; const float* b_att = (const float*)d_in[19];
    const float* W_ol  = (const float*)d_in[20]; const float* b_ol  = (const float*)d_in[21];
    const float* W_fc  = (const float*)d_in[22]; const float* b_fc  = (const float*)d_in[23];

    // workspace layout (all offsets 256B-aligned by construction)
    char* w = (char*)d_ws;
    __bf16* Wc    = (__bf16*)w;  w += (size_t)4 * HDIM * KTOT * 2;          // 3.0 MB
    float*  bc    = (float*)w;   w += (size_t)4 * HDIM * 4;
    __bf16* hs    = (__bf16*)w;  w += (size_t)TLEN * BSZ * HDIM * 2;        // 32 MB
    float*  hlast = (float*)w;   w += (size_t)BSZ * HDIM * 4;
    float*  fh    = (float*)w;   w += (size_t)BSZ * 256 * 4;
    float*  score = (float*)w;   w += (size_t)BSZ * HDIM * 4;
    float*  att   = (float*)w;   w += (size_t)BSZ * TLEN * 4;
    float*  dist  = (float*)w;   w += (size_t)BSZ * TLEN * 4;
    float*  attout = (float*)w;  w += (size_t)BSZ * HDIM * 4;

    float* out_fc    = (float*)d_out;          // [64,2] sigmoid outputs
    float* out_hlast = (float*)d_out + 128;    // [64,512] h_last

    // 1) fuse + convert weights/biases
    {
        int tot = 4 * HDIM * KTOT;
        k_prep<<<(tot + 255) / 256, 256, 0, stream>>>(
            Wir, Wif, Wig, Wio, Whr, Whf, Whg, Who,
            bir, bif, big_, bio, bhr, bhf, bhg, bho, Wc, bc);
    }

    // 2) persistent recurrent scan (single WGP, 32 waves, WMMA bf16)
    {
        size_t smem = (size_t)BSZ * EDIM * 2 + 2u * BSZ * HDIM * 2;  // 160 KB of 320 KB WGP LDS
        lstm_scan<<<1, 1024, smem, stream>>>(input_words, emb, Wc, bc, hs, hlast, out_hlast);
    }

    // 3) attention + output head
    k_fh<<<(BSZ * 256 + 255) / 256, 256, 0, stream>>>(hlast, W_ol, b_ol, fh);
    k_score<<<(BSZ * HDIM + 255) / 256, 256, 0, stream>>>(fh, W_att, b_att, score);
    k_attsc<<<(BSZ * TLEN + 255) / 256, 256, 0, stream>>>(score, hs, att);
    k_softmax<<<BSZ, 256, 0, stream>>>(att, dist);
    k_attout<<<(BSZ * HDIM + 255) / 256, 256, 0, stream>>>(hs, dist, attout);
    k_out<<<1, 128, 0, stream>>>(fh, attout, W_fc, b_fc, out_fc);
}
